// LevelReasoning_65867618451913
// MI455X (gfx1250) — matrix operand order repaired and checked by reading, not compiled, not run on hardware
//
#include <hip/hip_runtime.h>

typedef _Float16 h16;
typedef __attribute__((ext_vector_type(16))) _Float16 v16h;
typedef __attribute__((ext_vector_type(8)))  _Float16 v8h;
typedef __attribute__((ext_vector_type(8)))  float    v8f;

#define MROWS 114688L   // B*N = 16384*7
#define CDIM  512

// ---------------------------------------------------------------------------
// fp32 -> f16 conversion (weights, once per launch; cheap, deterministic)
// ---------------------------------------------------------------------------
__global__ __launch_bounds__(256) void cvt_f32_f16(const float* __restrict__ in,
                                                   h16* __restrict__ out, int n) {
    int i = blockIdx.x * 256 + threadIdx.x;
    if (i < n) out[i] = (h16)in[i];
}

__global__ __launch_bounds__(256) void zero_f32(float* __restrict__ p, int n) {
    int i = blockIdx.x * 256 + threadIdx.x;
    if (i < n) p[i] = 0.f;
}

// ---------------------------------------------------------------------------
// Core WMMA GEMM:  out[M, Nout] = A[M, 512] @ W[Nout, 512]^T + bias
// One wave -> one 16x64 output strip (4 accumulators sharing one A fragment),
// K-loop of 16 steps x 4 independent v_wmma_f32_16x16x32_f16, with fragments
// for step k+32 loaded before the WMMAs of step k (software double-buffer).
// Lane mappings used here (A/B: contiguous k, not the ISA table order):
//   A (16x32 f16):  M = lane&15 ; K = (lane>>4)*16 + e
//   B (32x16 f16):  N = lane&15 ; K = (lane>>4)*16 + e
//   C/D (16x16 f32): N = lane&15 ; M = (lane>>4)*8 + r
// ---------------------------------------------------------------------------
template<bool A_F32, bool OUT_F32>
__global__ __launch_bounds__(256) void gemm512_wmma(
    const void* __restrict__ A_, const h16* __restrict__ W,
    const float* __restrict__ bias, void* __restrict__ out_,
    int ngl2 /* log2(Nout/64) */, long Mrows)
{
    const int  wave  = threadIdx.x >> 5;
    const int  lane  = threadIdx.x & 31;
    const long wid   = (long)blockIdx.x * 8 + wave;
    const long mtile = wid >> ngl2;
    const int  ng    = (int)(wid & ((1 << ngl2) - 1));
    if (mtile * 16 >= Mrows) return;          // wave-uniform: EXEC stays all-1s

    const int  am    = lane & 15;
    const int  abase = (lane >> 4) * 16;
    const int  bn    = lane & 15;
    const int  bk    = (lane >> 4) * 16;
    const int  Nout  = 64 << ngl2;

    const long arow = mtile * 16 + am;
    const h16*   A16 = (const h16*)A_ + arow * CDIM;
    const float* A32 = (const float*)A_ + arow * CDIM;
    const h16*   W0  = W + (long)(ng * 64 + bn) * CDIM;  // tile t adds t*16*CDIM

    auto loadA = [&](int k0) -> v16h {
        v16h a;
        if (A_F32) {
            v8f lo = *(const v8f*)(A32 + k0 + abase);
            v8f hi = *(const v8f*)(A32 + k0 + abase + 8);
            #pragma unroll
            for (int j = 0; j < 8; ++j) { a[j] = (h16)lo[j]; a[j + 8] = (h16)hi[j]; }
        } else {
            v8h lo = *(const v8h*)(A16 + k0 + abase);
            v8h hi = *(const v8h*)(A16 + k0 + abase + 8);
            #pragma unroll
            for (int j = 0; j < 8; ++j) { a[j] = lo[j]; a[j + 8] = hi[j]; }
        }
        return a;
    };
    auto loadB = [&](int k0, int t) -> v16h {
        return *(const v16h*)(W0 + (long)t * 16 * CDIM + k0 + bk);  // 32B, aligned
    };

    v8f acc0 = {}, acc1 = {}, acc2 = {}, acc3 = {};

    v16h a  = loadA(0);
    v16h b0 = loadB(0, 0), b1 = loadB(0, 1), b2 = loadB(0, 2), b3 = loadB(0, 3);

    #pragma unroll
    for (int k0 = 0; k0 < CDIM; k0 += 32) {
        v16h an = {}, bn0 = {}, bn1 = {}, bn2 = {}, bn3 = {};
        if (k0 + 32 < CDIM) {                 // compile-time after full unroll
            an  = loadA(k0 + 32);
            bn0 = loadB(k0 + 32, 0);
            bn1 = loadB(k0 + 32, 1);
            bn2 = loadB(k0 + 32, 2);
            bn3 = loadB(k0 + 32, 3);
        }
        acc0 = __builtin_amdgcn_wmma_f32_16x16x32_f16(false, a, false, b0, (short)0, acc0, false, false);
        acc1 = __builtin_amdgcn_wmma_f32_16x16x32_f16(false, a, false, b1, (short)0, acc1, false, false);
        acc2 = __builtin_amdgcn_wmma_f32_16x16x32_f16(false, a, false, b2, (short)0, acc2, false, false);
        acc3 = __builtin_amdgcn_wmma_f32_16x16x32_f16(false, a, false, b3, (short)0, acc3, false, false);
        a = an; b0 = bn0; b1 = bn1; b2 = bn2; b3 = bn3;
    }

    const long rbase = mtile * 16 + (lane >> 4) * 8;
    auto store = [&](const v8f& acc, int t) {
        const int   col  = ng * 64 + t * 16 + bn;
        const float bval = bias[col];
        #pragma unroll
        for (int r = 0; r < 8; ++r) {
            float vo = acc[r] + bval;
            if (OUT_F32) ((float*)out_)[(rbase + r) * (long)Nout + col] = vo;
            else         ((h16*)  out_)[(rbase + r) * (long)Nout + col] = (h16)vo;
        }
    };
    store(acc0, 0); store(acc1, 1); store(acc2, 2); store(acc3, 3);
}

// ---------------------------------------------------------------------------
// BatchNorm: stats over all (B,N) rows per channel.  Each block reduces a
// 128-row chunk privately (coalesced 512B row reads), then 4 float atomics
// per thread into stats[0..511]=sum, stats[512..1023]=sumsq.
// ---------------------------------------------------------------------------
__global__ __launch_bounds__(256) void bn_stats(const h16* __restrict__ h,
                                                float* __restrict__ stats) {
    const int  t  = threadIdx.x;
    const long r0 = (long)blockIdx.x * 128;
    float s0 = 0.f, s1 = 0.f, q0 = 0.f, q1 = 0.f;
    for (int r = 0; r < 128; ++r) {
        const h16* rowp = h + (r0 + r) * CDIM;
        float a = (float)rowp[t];
        float b = (float)rowp[t + 256];
        s0 += a; q0 += a * a;
        s1 += b; q1 += b * b;
    }
    atomicAdd(&stats[t],             s0);
    atomicAdd(&stats[512 + t],       q0);
    atomicAdd(&stats[t + 256],       s1);
    atomicAdd(&stats[512 + t + 256], q1);
}

__global__ __launch_bounds__(256) void bn_finalize(const float* __restrict__ stats,
                                                   const float* __restrict__ g,
                                                   const float* __restrict__ b,
                                                   float* __restrict__ scsh, float invM) {
    int c = blockIdx.x * 256 + threadIdx.x;  // 512 channels
    float mean = stats[c] * invM;
    float var  = stats[512 + c] * invM - mean * mean;   // biased, matches ref
    float sc   = g[c] * rsqrtf(var + 1e-5f);
    scsh[c]       = sc;
    scsh[512 + c] = b[c] - mean * sc;
}

__global__ __launch_bounds__(256) void bn_apply(const h16* __restrict__ h,
                                                const float* __restrict__ scsh,
                                                h16* __restrict__ y, long n8) {
    long i = (long)blockIdx.x * 256 + threadIdx.x;
    if (i >= n8) return;
    int cg = (int)(i & 63) * 8;                 // 512/8 channel groups per row
    v8h hv = *(const v8h*)(h + i * 8);
    v8h ov;
    #pragma unroll
    for (int j = 0; j < 8; ++j)
        ov[j] = (h16)((float)hv[j] * scsh[cg + j] + scsh[512 + cg + j]);
    *(v8h*)(y + i * 8) = ov;
}

// ---------------------------------------------------------------------------
// Attention over the 7-node axis.  Tiny (~1 GFLOP total) -> VALU kernels.
// scores: one thread per (b,n): 7 dots of d=64, softmax over 7.
// ---------------------------------------------------------------------------
__global__ __launch_bounds__(256) void attn_scores(const h16* __restrict__ q,
                                                   const h16* __restrict__ k,
                                                   float* __restrict__ att, long rows) {
    long idx = (long)blockIdx.x * 256 + threadIdx.x;   // row = b*7 + n
    if (idx >= rows) return;
    long b = idx / 7;
    const h16* qp = q + idx * 64;
    float s[7], mx = -1e30f;
    #pragma unroll
    for (int m = 0; m < 7; ++m) {
        const h16* kp = k + (b * 7 + m) * 64;
        float acc = 0.f;
        #pragma unroll
        for (int d = 0; d < 64; ++d) acc += (float)qp[d] * (float)kp[d];
        s[m] = acc;
        mx = fmaxf(mx, acc);
    }
    float sum = 0.f;
    #pragma unroll
    for (int m = 0; m < 7; ++m) { s[m] = __expf(s[m] - mx); sum += s[m]; }
    float inv = 1.f / sum;
    #pragma unroll
    for (int m = 0; m < 7; ++m) att[idx * 7 + m] = s[m] * inv;
}

// out = relu(gam * (att @ v) + y)  -- one thread per (row, 8-channel group)
__global__ __launch_bounds__(256) void attn_av(const float* __restrict__ att,
                                               const h16* __restrict__ v,
                                               const h16* __restrict__ y,
                                               const float* __restrict__ gam,
                                               h16* __restrict__ out, long rows) {
    long idx = (long)blockIdx.x * 256 + threadIdx.x;
    long row = idx >> 6;
    int  cg  = (int)(idx & 63) * 8;
    if (row >= rows) return;
    long  b = row / 7;
    float g = gam[0];
    float a[7];
    #pragma unroll
    for (int m = 0; m < 7; ++m) a[m] = att[row * 7 + m];
    float acc[8] = {0.f, 0.f, 0.f, 0.f, 0.f, 0.f, 0.f, 0.f};
    #pragma unroll
    for (int m = 0; m < 7; ++m) {
        v8h vv = *(const v8h*)(v + (b * 7 + m) * CDIM + cg);
        #pragma unroll
        for (int j = 0; j < 8; ++j) acc[j] += a[m] * (float)vv[j];
    }
    v8h yy = *(const v8h*)(y + row * CDIM + cg);
    v8h ov;
    #pragma unroll
    for (int j = 0; j < 8; ++j)
        ov[j] = (h16)fmaxf(g * acc[j] + (float)yy[j], 0.f);
    *(v8h*)(out + row * CDIM + cg) = ov;
}

// ---------------------------------------------------------------------------
extern "C" void kernel_launch(void* const* d_in, const int* in_sizes, int n_in,
                              void* d_out, int out_size, void* d_ws, size_t ws_size,
                              hipStream_t stream) {
    const float* x  = (const float*)d_in[0];
    const float* Wt = (const float*)d_in[1];
    const float* bt = (const float*)d_in[2];
    const float *bng[3], *bnb[3], *Wq[3], *bq[3], *Wk[3], *bk[3], *Wv[3], *bv[3], *gam[3];
    for (int i = 0; i < 3; ++i) {
        int base = 3 + i * 9;
        bng[i] = (const float*)d_in[base + 0];
        bnb[i] = (const float*)d_in[base + 1];
        Wq[i]  = (const float*)d_in[base + 2];
        bq[i]  = (const float*)d_in[base + 3];
        Wk[i]  = (const float*)d_in[base + 4];
        bk[i]  = (const float*)d_in[base + 5];
        Wv[i]  = (const float*)d_in[base + 6];
        bv[i]  = (const float*)d_in[base + 7];
        gam[i] = (const float*)d_in[base + 8];
    }
    const float* Wb = (const float*)d_in[30];
    const float* bb = (const float*)d_in[31];

    // ---- carve workspace ----
    char* p = (char*)d_ws;
    auto carve = [&](size_t bytes) { void* r = p; p += (bytes + 255) & ~(size_t)255; return r; };
    h16*  hbuf  = (h16*)  carve(MROWS * CDIM * sizeof(h16));
    h16*  ybuf  = (h16*)  carve(MROWS * CDIM * sizeof(h16));
    h16*  vbuf  = (h16*)  carve(MROWS * CDIM * sizeof(h16));
    h16*  qbuf  = (h16*)  carve(MROWS * 64   * sizeof(h16));
    h16*  kbuf  = (h16*)  carve(MROWS * 64   * sizeof(h16));
    float* attb = (float*)carve(MROWS * 7    * sizeof(float));
    h16*  wWt   = (h16*)  carve(512 * 512 * sizeof(h16));
    h16*  wWb   = (h16*)  carve(256 * 512 * sizeof(h16));
    h16 *wWq[3], *wWk[3], *wWv[3];
    for (int i = 0; i < 3; ++i) {
        wWq[i] = (h16*)carve(64  * 512 * sizeof(h16));
        wWk[i] = (h16*)carve(64  * 512 * sizeof(h16));
        wWv[i] = (h16*)carve(512 * 512 * sizeof(h16));
    }
    float* stats = (float*)carve(1024 * sizeof(float));
    float* scsh  = (float*)carve(1024 * sizeof(float));

    // ---- convert weights to f16 (L2-resident, reused by every M-tile) ----
    cvt_f32_f16<<<1024, 256, 0, stream>>>(Wt, wWt, 512 * 512);
    cvt_f32_f16<<< 512, 256, 0, stream>>>(Wb, wWb, 256 * 512);
    for (int i = 0; i < 3; ++i) {
        cvt_f32_f16<<< 128, 256, 0, stream>>>(Wq[i], wWq[i], 64  * 512);
        cvt_f32_f16<<< 128, 256, 0, stream>>>(Wk[i], wWk[i], 64  * 512);
        cvt_f32_f16<<<1024, 256, 0, stream>>>(Wv[i], wWv[i], 512 * 512);
    }

    const long n8 = MROWS * CDIM / 8;   // 7,340,032 -> 28672 blocks of 256

    // ---- h = x @ Wt^T + bt  (fp32 A path, f16 out): 7168 mtiles x 8 ngroups ----
    gemm512_wmma<true, false><<<7168, 256, 0, stream>>>(
        (const void*)x, wWt, bt, (void*)hbuf, 3, MROWS);

    // ---- 3 x [BN -> q/k/v GEMMs -> attention -> residual+ReLU] ----
    for (int l = 0; l < 3; ++l) {
        zero_f32  <<<4, 256, 0, stream>>>(stats, 1024);
        bn_stats  <<<896, 256, 0, stream>>>(hbuf, stats);
        bn_finalize<<<2, 256, 0, stream>>>(stats, bng[l], bnb[l], scsh,
                                           1.0f / (float)MROWS);
        bn_apply  <<<28672, 256, 0, stream>>>(hbuf, scsh, ybuf, n8);

        gemm512_wmma<false, false><<< 896, 256, 0, stream>>>(
            (const void*)ybuf, wWq[l], bq[l], (void*)qbuf, 0, MROWS);
        gemm512_wmma<false, false><<< 896, 256, 0, stream>>>(
            (const void*)ybuf, wWk[l], bk[l], (void*)kbuf, 0, MROWS);
        gemm512_wmma<false, false><<<7168, 256, 0, stream>>>(
            (const void*)ybuf, wWv[l], bv[l], (void*)vbuf, 3, MROWS);

        attn_scores<<<  448, 256, 0, stream>>>(qbuf, kbuf, attb, MROWS);
        attn_av    <<<28672, 256, 0, stream>>>(attb, vbuf, ybuf, gam[l],
                                               hbuf, MROWS);
    }

    // ---- out = h @ Wb^T + bb  (fp32 out to d_out): 7168 x 4 ngroups ----
    gemm512_wmma<false, true><<<3584, 256, 0, stream>>>(
        (const void*)hbuf, wWb, bb, d_out, 2, MROWS);
}
